// LlamaSmoothquantMLP_51659866636889
// MI455X (gfx1250) — 8-bit matrix operand order repaired and checked by reading, compiled, not run on hardware
//
#include <hip/hip_runtime.h>
#include <hip/hip_bf16.h>
#include <math.h>
#include <stdint.h>

#define HIDDEN 4096
#define INTER  11008
#define NTOK   4096

typedef __attribute__((ext_vector_type(8))) int v8i;
typedef __attribute__((ext_vector_type(4))) int v4i;

#if defined(__AMDGCN__) && __has_builtin(__builtin_amdgcn_global_load_async_to_lds_b128)
#define USE_ASYNC 1
typedef __attribute__((address_space(1))) v4i as1_v4i;
typedef __attribute__((address_space(3))) v4i as3_v4i;
#define WAIT_ASYNC(n) asm volatile("s_wait_asynccnt " #n ::: "memory")
#else
#define USE_ASYNC 0
#define WAIT_ASYNC(n)
#endif

__device__ __forceinline__ void copy16(const signed char* g, signed char* l) {
#if USE_ASYNC
  __builtin_amdgcn_global_load_async_to_lds_b128(
      (as1_v4i*)(uintptr_t)g,
      (as3_v4i*)(unsigned int)(uintptr_t)l,
        0,   0);
#else
  *(int4*)l = *(const int4*)g;
#endif
}

__device__ __forceinline__ v8i wmma_iu8(v8i a, v8i b, v8i c) {
  return __builtin_amdgcn_wmma_i32_16x16x64_iu8(true, a, true, b, c, false, false);
}

__device__ __forceinline__ v8i zero8() {
  v8i z;
#pragma unroll
  for (int i = 0; i < 8; ++i) z[i] = 0;
  return z;
}

__device__ __forceinline__ v8i load_a_frag(const signed char* tile, int lane) {
  const int m  = lane & 15;
  const int kb = (lane >> 4) << 4;
  const signed char* p = tile + m * 64 + kb;
  v8i a;
#pragma unroll
  for (int j = 0; j < 4; ++j) {
    unsigned long long d = *(const unsigned long long*)(p + 8 * j + 16 * (j >> 1));
    a[2 * j]     = (int)(unsigned int)(d & 0xffffffffull);
    a[2 * j + 1] = (int)(unsigned int)(d >> 32);
  }
  return a;
}

__device__ __forceinline__ v8i load_b_frag(const signed char* tile, int lane) {
  const int n  = lane & 15;
  const int kb = (lane >> 4) << 4;
  const signed char* p = tile + n * 64 + kb;
  v4i lo = *(const v4i*)(p);
  v4i hi = *(const v4i*)(p + 32);
  v8i b;
#pragma unroll
  for (int j = 0; j < 4; ++j) { b[j] = lo[j]; b[j + 4] = hi[j]; }
  return b;
}

__global__ __launch_bounds__(256) void mlp_gateup_kernel(
    const signed char* __restrict__ x,
    const signed char* __restrict__ gW,
    const float* __restrict__ gB,
    const float* __restrict__ ga_p,
    const signed char* __restrict__ uW,
    const float* __restrict__ uB,
    const float* __restrict__ ua_p,
    const float* __restrict__ qs_p,
    signed char* __restrict__ interq)
{
  __shared__ __align__(16) signed char lsA[2][128 * 64];
  __shared__ __align__(16) signed char lsG[2][128 * 64];
  __shared__ __align__(16) signed char lsU[2][128 * 64];

  const int n0 = blockIdx.x * 128;
  const int m0 = blockIdx.y * 128;
  const int t = threadIdx.x;
  const int lane = t & 31;
  const int wave = t >> 5;
  const int waveM = wave & 3;
  const int waveN = wave >> 2;

  const float ga  = *ga_p;
  const float ua  = *ua_p;
  const float rqs = 1.0f / *qs_p;

  v8i accG[2][4], accU[2][4];
#pragma unroll
  for (int i = 0; i < 2; ++i)
#pragma unroll
    for (int j = 0; j < 4; ++j) { accG[i][j] = zero8(); accU[i][j] = zero8(); }

#define STAGE1(KB, BUF)                                                                \
  {                                                                                    \
    _Pragma("unroll")                                                                  \
    for (int c = 0; c < 2; ++c) {                                                      \
      const int chunk = t + c * 256;                                                   \
      const int row = chunk >> 2;                                                      \
      const int off = (chunk & 3) << 4;                                                \
      const int lin = chunk << 4;                                                      \
      copy16(x  + (size_t)(m0 + row) * HIDDEN + (KB) + off, &lsA[BUF][lin]);           \
      copy16(gW + (size_t)(n0 + row) * HIDDEN + (KB) + off, &lsG[BUF][lin]);           \
      copy16(uW + (size_t)(n0 + row) * HIDDEN + (KB) + off, &lsU[BUF][lin]);           \
    }                                                                                  \
  }

  const int KSTEPS = HIDDEN / 64;
  STAGE1(0, 0);

  for (int i = 0; i < KSTEPS; ++i) {
    const int cur = i & 1;
    if (i + 1 < KSTEPS) {
      STAGE1((i + 1) * 64, cur ^ 1);
      WAIT_ASYNC(0x6);
    } else {
      WAIT_ASYNC(0x0);
    }
    __syncthreads();

    const v8i a0 = load_a_frag(&lsA[cur][(waveM * 32 + 0)  * 64], lane);
    const v8i a1 = load_a_frag(&lsA[cur][(waveM * 32 + 16) * 64], lane);
#pragma unroll
    for (int nt = 0; nt < 4; ++nt) {
      const int boff = (waveN * 64 + nt * 16) * 64;
      const v8i bg = load_b_frag(&lsG[cur][boff], lane);
      accG[0][nt] = wmma_iu8(a0, bg, accG[0][nt]);
      accG[1][nt] = wmma_iu8(a1, bg, accG[1][nt]);
      const v8i bu = load_b_frag(&lsU[cur][boff], lane);
      accU[0][nt] = wmma_iu8(a0, bu, accU[0][nt]);
      accU[1][nt] = wmma_iu8(a1, bu, accU[1][nt]);
    }
    __syncthreads();
  }
#undef STAGE1

  const int lhalf = lane >> 4;
  const int lcol  = lane & 15;
#pragma unroll
  for (int mt = 0; mt < 2; ++mt) {
#pragma unroll
    for (int nt = 0; nt < 4; ++nt) {
      const int n = n0 + waveN * 64 + nt * 16 + lcol;
      const float gb = gB[n];
      const float ub = uB[n];
#pragma unroll
      for (int r = 0; r < 8; ++r) {
        const int m = m0 + waveM * 32 + mt * 16 + lhalf * 8 + r;
        const float g = ga * (float)accG[mt][nt][r] + gb;
        const float s = g / (1.0f + __expf(-g));
        const float u = ua * (float)accU[mt][nt][r] + ub;
        float v = s * u * rqs;
        v = fminf(fmaxf(rintf(v), -128.0f), 127.0f);
        interq[(size_t)m * INTER + n] = (signed char)v;
      }
    }
  }
}

__global__ __launch_bounds__(256) void mlp_down_kernel(
    const signed char* __restrict__ interq,
    const signed char* __restrict__ dW,
    const float* __restrict__ dB,
    const float* __restrict__ da_p,
    float* __restrict__ out)
{
  __shared__ __align__(16) signed char lsA[2][128 * 64];
  __shared__ __align__(16) signed char lsB[2][128 * 64];

  const int n0 = blockIdx.x * 128;
  const int m0 = blockIdx.y * 128;
  const int t = threadIdx.x;
  const int lane = t & 31;
  const int wave = t >> 5;
  const int waveM = wave & 3;
  const int waveN = wave >> 2;

  const float da = *da_p;

  v8i acc[2][4];
#pragma unroll
  for (int i = 0; i < 2; ++i)
#pragma unroll
    for (int j = 0; j < 4; ++j) acc[i][j] = zero8();

#define STAGE2(KB, BUF)                                                                \
  {                                                                                    \
    _Pragma("unroll")                                                                  \
    for (int c = 0; c < 2; ++c) {                                                      \
      const int chunk = t + c * 256;                                                   \
      const int row = chunk >> 2;                                                      \
      const int off = (chunk & 3) << 4;                                                \
      const int lin = chunk << 4;                                                      \
      copy16(interq + (size_t)(m0 + row) * INTER + (KB) + off, &lsA[BUF][lin]);        \
      copy16(dW     + (size_t)(n0 + row) * INTER + (KB) + off, &lsB[BUF][lin]);        \
    }                                                                                  \
  }

  const int KSTEPS = INTER / 64;
  STAGE2(0, 0);

  for (int i = 0; i < KSTEPS; ++i) {
    const int cur = i & 1;
    if (i + 1 < KSTEPS) {
      STAGE2((i + 1) * 64, cur ^ 1);
      WAIT_ASYNC(0x4);
    } else {
      WAIT_ASYNC(0x0);
    }
    __syncthreads();

    const v8i a0 = load_a_frag(&lsA[cur][(waveM * 32 + 0)  * 64], lane);
    const v8i a1 = load_a_frag(&lsA[cur][(waveM * 32 + 16) * 64], lane);
#pragma unroll
    for (int nt = 0; nt < 4; ++nt) {
      const v8i b = load_b_frag(&lsB[cur][(waveN * 64 + nt * 16) * 64], lane);
      acc[0][nt] = wmma_iu8(a0, b, acc[0][nt]);
      acc[1][nt] = wmma_iu8(a1, b, acc[1][nt]);
    }
    __syncthreads();
  }
#undef STAGE2

  const int lhalf = lane >> 4;
  const int lcol  = lane & 15;
#pragma unroll
  for (int mt = 0; mt < 2; ++mt) {
#pragma unroll
    for (int nt = 0; nt < 4; ++nt) {
      const int n = n0 + waveN * 64 + nt * 16 + lcol;
      const float b = dB[n];
#pragma unroll
      for (int r = 0; r < 8; ++r) {
        const int m = m0 + waveM * 32 + mt * 16 + lhalf * 8 + r;
        out[(size_t)m * HIDDEN + n] = da * (float)acc[mt][nt][r] + b;
      }
    }
  }
}

extern "C" void kernel_launch(void* const* d_in, const int* in_sizes, int n_in,
                              void* d_out, int out_size, void* d_ws, size_t ws_size,
                              hipStream_t stream) {
  const signed char* x  = (const signed char*)d_in[0];
  const signed char* gW = (const signed char*)d_in[1];
  const float*       gB = (const float*)d_in[2];
  const float*       ga = (const float*)d_in[3];
  const signed char* uW = (const signed char*)d_in[4];
  const float*       uB = (const float*)d_in[5];
  const float*       ua = (const float*)d_in[6];
  const signed char* dW = (const signed char*)d_in[7];
  const float*       dB = (const float*)d_in[8];
  const float*       da = (const float*)d_in[9];
  const float*       qs = (const float*)d_in[10];

  float* out = (float*)d_out;
  signed char* interq = (signed char*)d_ws;

  dim3 blk(256, 1, 1);
  dim3 g1(INTER / 128, NTOK / 128, 1);
  hipLaunchKernelGGL(mlp_gateup_kernel, g1, blk, 0, stream,
                     x, gW, gB, ga, uW, uB, ua, qs, interq);

  dim3 g2(HIDDEN / 128, NTOK / 128, 1);
  hipLaunchKernelGGL(mlp_down_kernel, g2, blk, 0, stream,
                     interq, dW, dB, da, out);
}
